// AmortLayer_70042326663696
// MI455X (gfx1250) — matrix operand order repaired and checked by reading, not compiled, not run on hardware
//
#include <hip/hip_runtime.h>
#include <hip/hip_bf16.h>

typedef __attribute__((ext_vector_type(16))) _Float16 v16h;
typedef __attribute__((ext_vector_type(8)))  float    v8f;
typedef unsigned int u32x4 __attribute__((ext_vector_type(4)));
typedef int          i32x4 __attribute__((ext_vector_type(4)));
typedef int          i32x8 __attribute__((ext_vector_type(8)));

#define S_DIM 4
#define N_DIM 32768
#define H_DIM 100
#define HP    128
#define K_DIM 256
#define OUT_U_TOTAL (S_DIM * N_DIM * K_DIM)

#if defined(__gfx1250__) && __has_builtin(__builtin_amdgcn_tensor_load_to_lds) && \
    __has_builtin(__builtin_amdgcn_s_wait_tensorcnt)
#define HAVE_TDM 1
#else
#define HAVE_TDM 0
#endif

// ---------------- ws layout ----------------
// w2pack : 16384 halves  [4 kt][8 nt][32 lane][16 h]   32768 B
// w3pack : 65536 halves  [4 kt][32 nt][32 lane][16 h] 131072 B
// acc    : S*K*5 floats  (UTLU00,01,11, UTLv0,1)       20480 B
// wvec   : S*K*2 floats                                  8192 B

// A-fragment K index for 16-bit 16x32 WMMA (CDNA5 ISA):
// lane group g = lane>>4, element h: h<8 -> K=g*8+h ; h>=8 -> K=16+g*8+(h-8)
__device__ __forceinline__ int a_klocal(int g, int h) {
  return (h < 8) ? (g * 8 + h) : (16 + g * 8 + (h - 8));
}

#if HAVE_TDM
// TDM: DMA a 2D tile of packed-W3 B-fragments into LDS.
//   4 rows (kt) x 512 halves, row stride 32 tiles (16384 halves).
// D# per CDNA5 ISA.
__device__ __forceinline__ void tdm_load_w3(const _Float16* gsrc, unsigned lds_byte_off) {
  unsigned long long ga = (unsigned long long)gsrc;
  u32x4 g0;
  g0[0] = 1u;                                            // count=1 (valid), no gather
  g0[1] = lds_byte_off;                                  // lds_addr
  g0[2] = (unsigned)(ga & 0xFFFFFFFFu);                  // global_addr[31:0]
  g0[3] = (unsigned)((ga >> 32) & 0x01FFFFFFu) | (2u << 30); // global_addr[56:32] | type=2
  i32x8 g1;
  g1[0] = 0x00010000;        // workgroup_mask=0, data_size=1 (2 bytes)
  g1[1] = (int)(512u << 16); // tensor_dim0 = 512 (bits 79:48, low part)
  g1[2] = (int)(4u << 16);   // tensor_dim1 = 4   (bits 111:80, low part)
  g1[3] = (int)(512u << 16); // tile_dim0 = 512   (bits 127:112)
  g1[4] = 4;                 // tile_dim1 = 4, tile_dim2 = 0
  g1[5] = 16384;             // tensor_dim0_stride = 16384 halves
  g1[6] = 0;                 // stride hi / tensor_dim1_stride (unused)
  g1[7] = 0;
  i32x4 gz = {0, 0, 0, 0};   // groups 2/3: unused (2D tensor)
#if defined(__clang_major__) && __clang_major__ >= 23
  i32x8 gz8 = {0, 0, 0, 0, 0, 0, 0, 0};
  __builtin_amdgcn_tensor_load_to_lds(g0, g1, gz, gz, gz8, 0);
#else
  __builtin_amdgcn_tensor_load_to_lds(g0, g1, gz, gz, 0);
#endif
}
#endif

// ---------------- kernel 0: pack weights to f16 B-fragments, zero acc ----
__global__ void prep_kernel(const float* __restrict__ W2, const float* __restrict__ W3,
                            _Float16* __restrict__ w2p, _Float16* __restrict__ w3p,
                            float* __restrict__ acc) {
  int e = blockIdx.x * blockDim.x + threadIdx.x;
  if (e < 16384) {
    int tile = e >> 9, r = e & 511;
    int lane = r >> 4, h = r & 15;
    int kt = tile >> 3, nt = tile & 7;
    int k = kt * 32 + ((h < 8) ? ((lane >> 4) * 8 + h) : (16 + (lane >> 4) * 8 + (h - 8)));
    int n = nt * 16 + (lane & 15);
    float v = (k < H_DIM && n < H_DIM) ? W2[k * H_DIM + n] : 0.0f;
    w2p[e] = (_Float16)v;
  } else if (e < 16384 + 65536) {
    int e2 = e - 16384;
    int tile = e2 >> 9, r = e2 & 511;
    int lane = r >> 4, h = r & 15;
    int kt = tile >> 5, nt = tile & 31;
    int k = kt * 32 + ((h < 8) ? ((lane >> 4) * 8 + h) : (16 + (lane >> 4) * 8 + (h - 8)));
    int n = nt * 16 + (lane & 15);
    float v = (k < H_DIM) ? W3[k * 512 + n] : 0.0f;
    w3p[e2] = (_Float16)v;
  } else if (e < 16384 + 65536 + 5120) {
    acc[e - 16384 - 65536] = 0.0f;
  }
}

// ---------------- kernel 1: fused MLP (WMMA+TDM) + exp + reduction -------
__global__ __launch_bounds__(128) void mlp_kernel(
    const float* __restrict__ U, const float* __restrict__ W1,
    const float* __restrict__ b1, const float* __restrict__ b2,
    const float* __restrict__ b3,
    const _Float16* __restrict__ w2p, const _Float16* __restrict__ w3p,
    float* __restrict__ acc) {
  // 32 KB region, time-multiplexed:
  //   phase A: h2 transpose staging  [4 waves][32 pts][128]
  //   phase B: W3 TDM double buffer  [2][8192 halves] (16 KB each)
  __shared__ __align__(64) _Float16 sbig[16384];
  __shared__ float sacc[K_DIM * 5];
  __shared__ float su[4][32][2];
  __shared__ float sb2[HP];
  __shared__ float sb3[512];
  __shared__ float sW1[2][HP];
  __shared__ float sb1[HP];

  const int tid  = threadIdx.x;
  const int wave = tid >> 5;
  const int lane = tid & 31;
  const int col  = lane & 15;
  const int g    = lane >> 4;
  const int s    = blockIdx.y;
  const int nbase = blockIdx.x * 128;

  if (tid < HP) {
    sW1[0][tid] = (tid < H_DIM) ? W1[tid] : 0.f;
    sW1[1][tid] = (tid < H_DIM) ? W1[H_DIM + tid] : 0.f;
    sb1[tid]    = (tid < H_DIM) ? b1[tid] : 0.f;
    sb2[tid]    = (tid < H_DIM) ? b2[tid] : 0.f;
  }
  for (int i = tid; i < 512; i += 128) sb3[i] = b3[i];
  for (int i = tid; i < K_DIM * 5; i += 128) sacc[i] = 0.f;
  {
    int p = (s * N_DIM + nbase + tid) * 2;
    su[tid >> 5][tid & 31][0] = U[p + 0];
    su[tid >> 5][tid & 31][1] = U[p + 1];
  }
  __syncthreads();

  // ---- layer 1 (VALU) directly into A-fragment layout ----
  v16h a1[2][4];
  #pragma unroll
  for (int rb = 0; rb < 2; ++rb) {
    int m = rb * 16 + col;
    float u0 = su[wave][m][0], u1 = su[wave][m][1];
    #pragma unroll
    for (int kt = 0; kt < 4; ++kt) {
      #pragma unroll
      for (int h = 0; h < 16; ++h) {
        int k = kt * 32 + a_klocal(g, h);
        float v = fmaxf(sb1[k] + u0 * sW1[0][k] + u1 * sW1[1][k], 0.f);
        a1[rb][kt][h] = (_Float16)v;   // padded k>=100 -> exactly 0
      }
    }
  }

  // ---- layer 2: h2 = relu(h1 @ W2 + b2) via WMMA ----
  #pragma unroll
  for (int nt = 0; nt < 8; ++nt) {
    #pragma unroll
    for (int rb = 0; rb < 2; ++rb) {
      float bias = sb2[nt * 16 + col];
      v8f c = {bias, bias, bias, bias, bias, bias, bias, bias};
      #pragma unroll
      for (int kt = 0; kt < 4; ++kt) {
        v16h b = ((const v16h*)w2p)[(kt * 8 + nt) * 32 + lane];
        c = __builtin_amdgcn_wmma_f32_16x16x32_f16(false, a1[rb][kt], false, b,
                                                   (short)0, c, false, false);
      }
      #pragma unroll
      for (int r = 0; r < 8; ++r) {
        int m = rb * 16 + g * 8 + r;                 // C layout: M = r + g*8
        sbig[wave * 4096 + m * 128 + nt * 16 + col] = (_Float16)fmaxf(c[r], 0.f);
      }
    }
  }
  __syncthreads();

  // ---- transpose h2 from C layout back to A-fragment layout ----
  v16h a2[2][4];
  #pragma unroll
  for (int rb = 0; rb < 2; ++rb) {
    int m = rb * 16 + col;
    #pragma unroll
    for (int kt = 0; kt < 4; ++kt) {
      #pragma unroll
      for (int h = 0; h < 16; ++h) {
        a2[rb][kt][h] = sbig[wave * 4096 + m * 128 + kt * 32 + a_klocal(g, h)];
      }
    }
  }
  __syncthreads();   // everyone done with h2 region before TDM overwrites it

#if HAVE_TDM
  unsigned lds_base = (unsigned)(unsigned long long)(void*)sbig;
  if (wave == 0) {   // prologue: DMA j=0 tiles into buffer 0
    tdm_load_w3(w3p + 0 * 512,  lds_base);          // nt = 0   (prec cols)
    tdm_load_w3(w3p + 16 * 512, lds_base + 8192);   // nt = 16  (mu cols)
  }
#endif

  // ---- layer 3 (WMMA) + exp + UTLU/UTLv accumulation ----
  for (int j = 0; j < 16; ++j) {
#if HAVE_TDM
    if (wave == 0) __builtin_amdgcn_s_wait_tensorcnt(0);  // buffer j ready
    __syncthreads();                                      // publish; j-1 readers done
    if (wave == 0 && j < 15) {                            // DMA j+1 while computing j
      unsigned b = lds_base + (unsigned)(((j + 1) & 1) * 16384);
      tdm_load_w3(w3p + (j + 1) * 512,  b);
      tdm_load_w3(w3p + (j + 17) * 512, b + 8192);
    }
    const _Float16* wlo = sbig + (j & 1) * 8192;  // nt=j    tiles, kt-major
    const _Float16* whi = wlo + 4096;             // nt=j+16 tiles
#endif
    float t00 = 0.f, t01 = 0.f, t11 = 0.f, tv0 = 0.f, tv1 = 0.f;
    #pragma unroll
    for (int rb = 0; rb < 2; ++rb) {
      float bp = sb3[j * 16 + col];
      float bm = sb3[256 + j * 16 + col];
      v8f cp = {bp, bp, bp, bp, bp, bp, bp, bp};
      v8f cm = {bm, bm, bm, bm, bm, bm, bm, bm};
      #pragma unroll
      for (int kt = 0; kt < 4; ++kt) {
        v16h A = a2[rb][kt];
#if HAVE_TDM
        v16h Bp = ((const v16h*)wlo)[kt * 32 + lane];
        v16h Bm = ((const v16h*)whi)[kt * 32 + lane];
#else
        v16h Bp = ((const v16h*)w3p)[(kt * 32 + j) * 32 + lane];
        v16h Bm = ((const v16h*)w3p)[(kt * 32 + j + 16) * 32 + lane];
#endif
        cp = __builtin_amdgcn_wmma_f32_16x16x32_f16(false, A, false, Bp,
                                                    (short)0, cp, false, false);
        cm = __builtin_amdgcn_wmma_f32_16x16x32_f16(false, A, false, Bm,
                                                    (short)0, cm, false, false);
      }
      #pragma unroll
      for (int r = 0; r < 8; ++r) {
        int m = rb * 16 + g * 8 + r;
        float u0 = su[wave][m][0], u1 = su[wave][m][1];
        float p  = __expf(cp[r]);
        float mu = cm[r];
        t00 += p * u0 * u0;  t01 += p * u0 * u1;  t11 += p * u1 * u1;
        tv0 += p * mu * u0;  tv1 += p * mu * u1;
      }
    }
    int k = j * 16 + col;
    atomicAdd(&sacc[k * 5 + 0], t00);
    atomicAdd(&sacc[k * 5 + 1], t01);
    atomicAdd(&sacc[k * 5 + 2], t11);
    atomicAdd(&sacc[k * 5 + 3], tv0);
    atomicAdd(&sacc[k * 5 + 4], tv1);
  }
  __syncthreads();

  float* ga = acc + s * K_DIM * 5;
  for (int i = tid; i < K_DIM * 5; i += 128) atomicAdd(&ga[i], sacc[i]);
}

// ---------------- kernel 2: per-(s,k) 2x2 solve + KL reduction ----------
__global__ __launch_bounds__(256) void solve_kernel(
    const float* __restrict__ acc, const float* __restrict__ eps,
    float* __restrict__ wvec, float* __restrict__ out) {
  int s = blockIdx.x, k = threadIdx.x;
  const float* a = acc + (s * K_DIM + k) * 5;
  float a00 = a[0] + 1.0f, a01 = a[1], a11 = a[2] + 1.0f;   // q_prec (PRIOR_VAR=1)
  float v0 = a[3], v1 = a[4];
  float l00 = sqrtf(a00), l10 = a01 / l00, l11 = sqrtf(a11 - l10 * l10);
  float det = a00 * a11 - a01 * a01;
  float c00 = a11 / det, c01 = -a01 / det, c11 = a00 / det;  // q_cov
  float qm0 = c00 * v0 + c01 * v1, qm1 = c01 * v0 + c11 * v1;
  float lc00 = sqrtf(c00), lc10 = c01 / lc00, lc11 = sqrtf(c11 - lc10 * lc10);
  float e0 = eps[(s * K_DIM + k) * 2 + 0], e1 = eps[(s * K_DIM + k) * 2 + 1];
  wvec[(s * K_DIM + k) * 2 + 0] = qm0 + lc00 * e0;
  wvec[(s * K_DIM + k) * 2 + 1] = qm1 + lc10 * e0 + lc11 * e1;
  float logdet_cov = -2.f * (__logf(l00) + __logf(l11));
  float kl = 0.5f * (c00 + c11 + qm0 * qm0 + qm1 * qm1 - 2.f - logdet_cov);

  __shared__ float red[K_DIM];
  red[k] = kl;
  __syncthreads();
  for (int off = 128; off > 0; off >>= 1) {
    if (k < off) red[k] += red[k + off];
    __syncthreads();
  }
  if (k == 0) out[OUT_U_TOTAL + s] = red[0];
}

// ---------------- kernel 3: out_U = relu(U . w), store-bandwidth bound ---
__global__ __launch_bounds__(256) void outu_kernel(
    const float* __restrict__ U, const float* __restrict__ wvec,
    float4* __restrict__ out) {
  int gid = blockIdx.x * blockDim.x + threadIdx.x;
  int s   = gid >> 21;                               // N*64 = 2^21
  int rem = gid & ((1 << 21) - 1);
  int n   = rem >> 6;
  int kc  = rem & 63;
  float u0 = U[(s * N_DIM + n) * 2 + 0];
  float u1 = U[(s * N_DIM + n) * 2 + 1];
  const float* w = wvec + (s * K_DIM + kc * 4) * 2;
  float4 o;
  o.x = fmaxf(u0 * w[0] + u1 * w[1], 0.f);
  o.y = fmaxf(u0 * w[2] + u1 * w[3], 0.f);
  o.z = fmaxf(u0 * w[4] + u1 * w[5], 0.f);
  o.w = fmaxf(u0 * w[6] + u1 * w[7], 0.f);
  out[gid] = o;
}

extern "C" void kernel_launch(void* const* d_in, const int* in_sizes, int n_in,
                              void* d_out, int out_size, void* d_ws, size_t ws_size,
                              hipStream_t stream) {
  const float* U   = (const float*)d_in[0];
  const float* eps = (const float*)d_in[1];
  const float* W1  = (const float*)d_in[2];
  const float* b1  = (const float*)d_in[3];
  const float* W2  = (const float*)d_in[4];
  const float* b2  = (const float*)d_in[5];
  const float* W3  = (const float*)d_in[6];
  const float* b3  = (const float*)d_in[7];
  float* out = (float*)d_out;

  _Float16* w2p = (_Float16*)d_ws;         // 16384 halves
  _Float16* w3p = w2p + 16384;             // 65536 halves
  float*    acc = (float*)(w3p + 65536);   // 5120 floats
  float*    wv  = acc + 5120;              // 2048 floats

  {
    int total = 16384 + 65536 + 5120;
    prep_kernel<<<(total + 255) / 256, 256, 0, stream>>>(W2, W3, w2p, w3p, acc);
  }
  {
    dim3 grid(N_DIM / 128, S_DIM);
    mlp_kernel<<<grid, 128, 0, stream>>>(U, W1, b1, b2, b3, w2p, w3p, acc);
  }
  solve_kernel<<<S_DIM, K_DIM, 0, stream>>>(acc, eps, wv, out);
  {
    int total4 = OUT_U_TOTAL / 4;
    outu_kernel<<<total4 / 256, 256, 0, stream>>>(U, wv, (float4*)out);
  }
}
